// RepLKBlock_74182675137219
// MI455X (gfx1250) — matrix operand order repaired and checked by reading, not compiled, not run on hardware
//
#include <hip/hip_runtime.h>
#include <hip/hip_bf16.h>

#define EPSV 1e-5f
constexpr int BB = 32, CC = 256, HH = 512, LL = 4096, KW = 31, PADW = 15;

typedef __bf16  v16bf __attribute__((ext_vector_type(16)));
typedef float   v8f   __attribute__((ext_vector_type(8)));

union FragAB { v16bf v; uint4 q[2]; };

__device__ __forceinline__ unsigned short f2bf(float f) {
    union { float f; unsigned int u; } x; x.f = f;
    unsigned int r = x.u + 0x7FFFu + ((x.u >> 16) & 1u);
    return (unsigned short)(r >> 16);
}
__device__ __forceinline__ float bf2f(unsigned short h) {
    union { unsigned int u; float f; } x; x.u = ((unsigned int)h) << 16;
    return x.f;
}

// ---------------- precompute: fold BN0/BN1/BN2 into expand weights / dw weights ----
__global__ __launch_bounds__(256) void prep_expand(
    const float* __restrict__ wexp, const float* __restrict__ wdw,
    const float* __restrict__ g0, const float* __restrict__ b0,
    const float* __restrict__ m0, const float* __restrict__ v0,
    const float* __restrict__ g1, const float* __restrict__ b1,
    const float* __restrict__ m1, const float* __restrict__ v1,
    const float* __restrict__ g2, const float* __restrict__ b2,
    const float* __restrict__ m2, const float* __restrict__ v2,
    unsigned short* __restrict__ wexp_bf, float* __restrict__ bias1,
    float* __restrict__ wdwp, float* __restrict__ bias2)
{
    __shared__ float red[256];
    int h = blockIdx.x, c = threadIdx.x;
    float s0 = g0[c] * rsqrtf(v0[c] + EPSV);
    float t0 = b0[c] - m0[c] * s0;
    float s1 = g1[h] * rsqrtf(v1[h] + EPSV);
    float w  = wexp[h * CC + c];
    wexp_bf[h * CC + c] = f2bf(s1 * s0 * w);
    red[c] = w * t0;
    __syncthreads();
    for (int s = 128; s > 0; s >>= 1) { if (c < s) red[c] += red[c + s]; __syncthreads(); }
    float s2 = g2[h] * rsqrtf(v2[h] + EPSV);
    if (c < KW) wdwp[h * KW + c] = s2 * wdw[h * KW + c];
    if (c == 0) {
        bias1[h] = s1 * red[0] + (b1[h] - m1[h] * s1);
        bias2[h] = b2[h] - m2[h] * s2;
    }
}

// ---------------- precompute: fold BN3 into project weights -----------------------
__global__ __launch_bounds__(256) void prep_project(
    const float* __restrict__ wproj,
    const float* __restrict__ g3, const float* __restrict__ b3,
    const float* __restrict__ m3, const float* __restrict__ v3,
    unsigned short* __restrict__ wproj_bf, float* __restrict__ bias3)
{
    int c = blockIdx.x, t = threadIdx.x;
    float s3 = g3[c] * rsqrtf(v3[c] + EPSV);
    wproj_bf[c * HH + t]       = f2bf(s3 * wproj[c * HH + t]);
    wproj_bf[c * HH + t + 256] = f2bf(s3 * wproj[c * HH + t + 256]);
    if (t == 0) bias3[c] = b3[c] - m3[c] * s3;
}

// ---------------- GEMM1: h1 = relu(W'' @ x + bias1), bf16 out ---------------------
// Block: batch b, 32 L-columns; 8 waves, each wave 64 rows of H=512.
__global__ __launch_bounds__(256) void gemm_expand(
    const float* __restrict__ x, const unsigned short* __restrict__ wA,
    const float* __restrict__ bias1, unsigned short* __restrict__ h1)
{
    constexpr int LDSK = CC + 8; // 264 bf16 row stride, 16B-aligned rows
    __shared__ __align__(16) unsigned short xs[32 * LDSK];
    int b = blockIdx.y, l0 = blockIdx.x * 32;
    int tid = threadIdx.x, lane = tid & 31, w = tid >> 5;

    { // stage x tile [K=256][N=32] into LDS as K-contiguous per column n
        int n = tid & 31, crow = tid >> 5;
        const float* xb = x + ((size_t)b * CC) * LL + l0 + n;
        for (int rep = 0; rep < 32; ++rep) {
            int c = rep * 8 + crow;
            xs[n * LDSK + c] = f2bf(xb[(size_t)c * LL]);
        }
    }
    __syncthreads();

    int kg = lane >> 4, ln = lane & 15;
    v8f z = {0.f, 0.f, 0.f, 0.f, 0.f, 0.f, 0.f, 0.f};
    v8f acc[4][2];
    for (int rt = 0; rt < 4; ++rt) for (int ct = 0; ct < 2; ++ct) acc[rt][ct] = z;

    for (int k0 = 0; k0 < CC; k0 += 32) {
        FragAB bfr[2];
        int kb = k0 + kg * 16;
        #pragma unroll
        for (int ct = 0; ct < 2; ++ct) {
            const uint4* p = (const uint4*)&xs[(ct * 16 + ln) * LDSK + kb];
            bfr[ct].q[0] = p[0]; bfr[ct].q[1] = p[1];
        }
        #pragma unroll
        for (int rt = 0; rt < 4; ++rt) {
            int row = w * 64 + rt * 16 + ln;
            const unsigned short* ap = wA + (size_t)row * CC;
            __builtin_prefetch(ap + k0 + 32, 0, 3); // next A panel -> global_prefetch_b8
            FragAB af;
            af.q[0] = *(const uint4*)(ap + k0 + kg * 16);
            af.q[1] = *(const uint4*)(ap + k0 + kg * 16 + 8);
            acc[rt][0] = __builtin_amdgcn_wmma_f32_16x16x32_bf16(
                false, af.v, false, bfr[0].v, (short)0, acc[rt][0], false, false);
            acc[rt][1] = __builtin_amdgcn_wmma_f32_16x16x32_bf16(
                false, af.v, false, bfr[1].v, (short)0, acc[rt][1], false, false);
        }
    }

    #pragma unroll
    for (int rt = 0; rt < 4; ++rt) {
        int rowb = w * 64 + rt * 16 + kg * 8;
        #pragma unroll
        for (int ct = 0; ct < 2; ++ct) {
            int col = l0 + ct * 16 + ln;
            #pragma unroll
            for (int r = 0; r < 8; ++r) {
                int row = rowb + r;
                float vv = acc[rt][ct][r] + bias1[row];
                vv = vv > 0.f ? vv : 0.f;
                h1[((size_t)b * HH + row) * LL + col] = f2bf(vv);
            }
        }
    }
}

// ---------------- depthwise 31-tap conv + BN2 + ReLU ------------------------------
// Row staged global->LDS with gfx1250 async-to-LDS copies (ASYNCcnt path).
__global__ __launch_bounds__(256) void dwconv(
    const unsigned short* __restrict__ h1, const float* __restrict__ wdwp,
    const float* __restrict__ bias2, unsigned short* __restrict__ hdw)
{
    constexpr int OFF = 16; // halo shifted to keep async B128 dest 16B-aligned
    __shared__ __align__(16) unsigned short buf[OFF + LL + OFF]; // 4128
    __shared__ float wk[KW];
    __shared__ float bsh;
    int h = blockIdx.x, b = blockIdx.y, tid = threadIdx.x;
    const unsigned short* src = h1 + ((size_t)b * HH + h) * LL;

    // zero the halos with plain DS stores
    if (tid < OFF) buf[tid] = 0;
    else if (tid < 2 * OFF) buf[OFF + LL + (tid - OFF)] = 0;

    // async copy main body: 4096 bf16 = 512 x 16B chunks, 2 chunks/thread.
    // Flat LDS addressing truncates to addr[31:0] (ISA 10.2), so the low 32
    // bits of the generic shared pointer are the LDS byte address.
    {
        unsigned lbase = (unsigned)(size_t)(&buf[OFF]);
        #pragma unroll
        for (int rep = 0; rep < 2; ++rep) {
            int chunk = rep * 256 + tid;
            unsigned laddr = lbase + chunk * 16;
            const void* g = (const void*)(src + chunk * 8);
            asm volatile("global_load_async_to_lds_b128 %0, %1, off"
                         :: "v"(laddr), "v"(g) : "memory");
        }
    }
    if (tid < KW) wk[tid] = wdwp[h * KW + tid];
    if (tid == KW) bsh = bias2[h];
    asm volatile("s_wait_asynccnt 0" ::: "memory");
    __syncthreads();

    float bias = bsh;
    unsigned short* dst = hdw + ((size_t)b * HH + h) * LL;
    for (int j = 0; j < 16; ++j) {
        int l = j * 256 + tid;
        float s = bias;
        #pragma unroll
        for (int k = 0; k < KW; ++k) s += wk[k] * bf2f(buf[l + k + 1]); // buf[l+OFF] = src[l]
        s = s > 0.f ? s : 0.f;
        dst[l] = f2bf(s);
    }
}

// ---------------- GEMM2: out = x + Wp' @ hdw + bias3, fp32 out --------------------
__global__ __launch_bounds__(256) void gemm_project(
    const unsigned short* __restrict__ hdw, const unsigned short* __restrict__ wA,
    const float* __restrict__ bias3, const float* __restrict__ x,
    float* __restrict__ out)
{
    constexpr int LDSK = HH + 8; // 520 bf16 row stride
    __shared__ __align__(16) unsigned short hs[32 * LDSK];
    int b = blockIdx.y, l0 = blockIdx.x * 32;
    int tid = threadIdx.x, lane = tid & 31, w = tid >> 5;

    {
        int n = tid & 31, hrow = tid >> 5;
        const unsigned short* hb = hdw + ((size_t)b * HH) * LL + l0 + n;
        for (int rep = 0; rep < 64; ++rep) {
            int h = rep * 8 + hrow;
            hs[n * LDSK + h] = hb[(size_t)h * LL];
        }
    }
    __syncthreads();

    int kg = lane >> 4, ln = lane & 15;
    v8f z = {0.f, 0.f, 0.f, 0.f, 0.f, 0.f, 0.f, 0.f};
    v8f acc[2][2];
    for (int rt = 0; rt < 2; ++rt) for (int ct = 0; ct < 2; ++ct) acc[rt][ct] = z;

    for (int k0 = 0; k0 < HH; k0 += 32) {
        FragAB bfr[2];
        int kb = k0 + kg * 16;
        #pragma unroll
        for (int ct = 0; ct < 2; ++ct) {
            const uint4* p = (const uint4*)&hs[(ct * 16 + ln) * LDSK + kb];
            bfr[ct].q[0] = p[0]; bfr[ct].q[1] = p[1];
        }
        #pragma unroll
        for (int rt = 0; rt < 2; ++rt) {
            int row = w * 32 + rt * 16 + ln;
            const unsigned short* ap = wA + (size_t)row * HH;
            __builtin_prefetch(ap + k0 + 32, 0, 3);
            FragAB af;
            af.q[0] = *(const uint4*)(ap + k0 + kg * 16);
            af.q[1] = *(const uint4*)(ap + k0 + kg * 16 + 8);
            acc[rt][0] = __builtin_amdgcn_wmma_f32_16x16x32_bf16(
                false, af.v, false, bfr[0].v, (short)0, acc[rt][0], false, false);
            acc[rt][1] = __builtin_amdgcn_wmma_f32_16x16x32_bf16(
                false, af.v, false, bfr[1].v, (short)0, acc[rt][1], false, false);
        }
    }

    #pragma unroll
    for (int rt = 0; rt < 2; ++rt) {
        int rowb = w * 32 + rt * 16 + kg * 8;
        #pragma unroll
        for (int ct = 0; ct < 2; ++ct) {
            int col = l0 + ct * 16 + ln;
            #pragma unroll
            for (int r = 0; r < 8; ++r) {
                int row = rowb + r;
                size_t idx = ((size_t)b * CC + row) * LL + col;
                out[idx] = x[idx] + bias3[row] + acc[rt][ct][r];
            }
        }
    }
}

extern "C" void kernel_launch(void* const* d_in, const int* in_sizes, int n_in,
                              void* d_out, int out_size, void* d_ws, size_t ws_size,
                              hipStream_t stream) {
    const float* x    = (const float*)d_in[0];
    const float* wexp = (const float*)d_in[1];
    const float* wdw  = (const float*)d_in[2];
    const float* wprj = (const float*)d_in[3];
    const float *g0 = (const float*)d_in[4],  *b0 = (const float*)d_in[5],
                *m0 = (const float*)d_in[6],  *v0 = (const float*)d_in[7];
    const float *g1 = (const float*)d_in[8],  *b1 = (const float*)d_in[9],
                *m1 = (const float*)d_in[10], *v1 = (const float*)d_in[11];
    const float *g2 = (const float*)d_in[12], *b2 = (const float*)d_in[13],
                *m2 = (const float*)d_in[14], *v2 = (const float*)d_in[15];
    const float *g3 = (const float*)d_in[16], *b3 = (const float*)d_in[17],
                *m3 = (const float*)d_in[18], *v3 = (const float*)d_in[19];
    float* out = (float*)d_out;

    // workspace carve-up (256B aligned)
    char* ws = (char*)d_ws;
    size_t off = 0;
    unsigned short* h1  = (unsigned short*)(ws + off); off += (size_t)BB * HH * LL * 2; // 134 MB
    unsigned short* hdw = (unsigned short*)(ws + off); off += (size_t)BB * HH * LL * 2; // 134 MB
    unsigned short* wexp_bf = (unsigned short*)(ws + off); off += (size_t)HH * CC * 2;
    unsigned short* wprj_bf = (unsigned short*)(ws + off); off += (size_t)CC * HH * 2;
    float* wdwp  = (float*)(ws + off); off += ((size_t)HH * KW * 4 + 255) & ~(size_t)255;
    float* bias1 = (float*)(ws + off); off += 2048;
    float* bias2 = (float*)(ws + off); off += 2048;
    float* bias3 = (float*)(ws + off); off += 1024;
    (void)ws_size; (void)in_sizes; (void)n_in; (void)out_size;

    prep_expand<<<HH, 256, 0, stream>>>(wexp, wdw,
        g0, b0, m0, v0, g1, b1, m1, v1, g2, b2, m2, v2,
        wexp_bf, bias1, wdwp, bias2);
    prep_project<<<CC, 256, 0, stream>>>(wprj, g3, b3, m3, v3, wprj_bf, bias3);

    gemm_expand<<<dim3(LL / 32, BB), 256, 0, stream>>>(x, wexp_bf, bias1, h1);
    dwconv<<<dim3(HH, BB), 256, 0, stream>>>(h1, wdwp, bias2, hdw);
    gemm_project<<<dim3(LL / 32, BB), 256, 0, stream>>>(hdw, wprj_bf, bias3, x, out);
}
